// AttnBlock_52948356825623
// MI455X (gfx1250) — matrix operand order repaired and checked by reading, not compiled, not run on hardware
//
#include <hip/hip_runtime.h>
#include <hip/hip_bf16.h>
#include <math.h>

typedef __attribute__((ext_vector_type(16))) _Float16 v16h;
typedef __attribute__((ext_vector_type(8)))  float    v8f;
typedef __attribute__((ext_vector_type(4)))  float    v4f;

#define BATCH   2
#define CH      256
#define HW      4096
#define NHEADS  4
#define DH      64
#define KNN     64
#define NGROUPS 32
#define CPG     8
#define GEPS    1e-6f

#define KC_TILES 8     // 256 / 32 (K-dim tiles per GEMM)
#define NT_TILES 256   // 4096 / 16
#define MO_TILES 16    // 256 / 16
#define NTW      4     // N-tiles per wave (A-fragment reuse factor)

// ---------------------------------------------------------------------------
// Workspace layout (bytes)
// ---------------------------------------------------------------------------
#define HPACK_OFF   ((size_t)0)                        // f16 GN output, B-fragment layout: 4 MB
#define HPACK_BYTES ((size_t)BATCH*CH*HW*2)
#define APACK_OFF   (HPACK_OFF + HPACK_BYTES)          // f16 wq/wk/wv/wo, A-fragment layout: 512 KB
#define APACK_BYTES ((size_t)4*CH*CH*2)
#define QT_OFF      (APACK_OFF + APACK_BYTES)          // f32 transposed [b,h,n,c]: 8 MB each
#define ONE_T_BYTES ((size_t)BATCH*NHEADS*HW*DH*4)
#define KT_OFF      (QT_OFF + ONE_T_BYTES)
#define VT_OFF      (KT_OFF + ONE_T_BYTES)
#define ATTNP_OFF   (VT_OFF + ONE_T_BYTES)             // f16 attn, B-fragment layout: 4 MB

// ---------------------------------------------------------------------------
// Fragment-layout helpers (wave32 WMMA 16x16x32 f16, contiguous k = 16*(lane>>4) + j for A and B)
//   A (16x32, MxK): lane = ((kk>>4)<<4)|M, j = kk&15
//   B (32x16, KxN): lane = ((kk>>4)<<4)|nlo, j = kk&15
//   D (16x16 f32):  VGPR r, lanes0-15 -> M=r, lanes16-31 -> M=r+8; N = lane&15
// ---------------------------------------------------------------------------

// ---------------------------------------------------------------------------
// 1) GroupNorm + pack into f16 B-fragment layout.
//    One block per (batch, group): reduce 8*4096 values, normalize, pack.
// ---------------------------------------------------------------------------
__global__ __launch_bounds__(1024) void gn_pack_kernel(
    const float* __restrict__ x, const float* __restrict__ scale,
    const float* __restrict__ bias, _Float16* __restrict__ hpack) {
  __shared__ float ssum[1024];
  __shared__ float ssq[1024];
  const int b  = blockIdx.x >> 5;
  const int g  = blockIdx.x & 31;
  const int c0 = g * CPG;
  const int NEL = CPG * HW;  // 32768
  const int tid = threadIdx.x;
  const float* xb = x + ((size_t)b * CH + c0) * HW;

  float s = 0.f, s2 = 0.f;
  for (int i = tid; i < NEL; i += 1024) {
    float v = xb[i];
    s += v; s2 += v * v;
  }
  ssum[tid] = s; ssq[tid] = s2;
  __syncthreads();
  for (int off = 512; off > 0; off >>= 1) {
    if (tid < off) { ssum[tid] += ssum[tid + off]; ssq[tid] += ssq[tid + off]; }
    __syncthreads();
  }
  const float mean = ssum[0] * (1.f / NEL);
  const float var  = ssq[0] * (1.f / NEL) - mean * mean;
  const float rstd = rsqrtf(var + GEPS);

  for (int i = tid; i < NEL; i += 1024) {
    const int c = c0 + (i >> 12);
    const int n = i & 4095;
    const float v = (xb[i] - mean) * rstd * scale[c] + bias[c];
    const int kc = c >> 5, cr = c & 31;
    const int lane = ((cr >> 4) << 4) | (n & 15);
    const int j = cr & 15;
    const int nt = n >> 4;
    hpack[((((size_t)b * KC_TILES + kc) * NT_TILES + nt) * 32 + lane) * 16 + j] =
        (_Float16)v;
  }
}

// ---------------------------------------------------------------------------
// 2) Pack the four weight matrices into f16 A-fragment layout.
// ---------------------------------------------------------------------------
__global__ __launch_bounds__(256) void pack_w_kernel(
    const float* __restrict__ wq, const float* __restrict__ wk,
    const float* __restrict__ wv, const float* __restrict__ wo,
    _Float16* __restrict__ apack) {
  const int t = blockIdx.x * 256 + threadIdx.x;   // 4 * 65536 threads total
  const int mat = t >> 16;
  const int e = t & 65535;
  const int o = e >> 8, c = e & 255;
  const float* w = (mat == 0) ? wq : (mat == 1) ? wk : (mat == 2) ? wv : wo;
  const float v = w[o * CH + c];
  const int mo = o >> 4, M = o & 15;
  const int kc = c >> 5, kk = c & 31;
  const int blk = kk >> 3;
  const int lane = ((kk >> 4) << 4) | M;
  const int j = kk & 15;
  apack[((((size_t)mat * MO_TILES + mo) * KC_TILES + kc) * 32 + lane) * 16 + j] =
      (_Float16)v;
}

// ---------------------------------------------------------------------------
// 3) Q/K/V GEMM: one wave per 16x64 output strip (1 mo x 4 nt tiles).
//    A fragment loaded once per kc step and reused by 4 WMMAs into 4
//    independent accumulators (deep XDL pipelining, 4x less A traffic).
//    Output stored transposed per head: [b, head, n, c] (256B rows) so the
//    attention gathers read whole cachelines.
// ---------------------------------------------------------------------------
__global__ __launch_bounds__(256) void qkv_gemm_kernel(
    const _Float16* __restrict__ apack, const _Float16* __restrict__ hpack,
    const float* __restrict__ bq, const float* __restrict__ bk,
    const float* __restrict__ bv,
    float* __restrict__ qt, float* __restrict__ kt, float* __restrict__ vt) {
  const int lane = threadIdx.x & 31;
  const int w = (blockIdx.x << 3) + (threadIdx.x >> 5);
  const int nt0 = (w & (NT_TILES / NTW - 1)) * NTW;   // base of 4-tile strip
  int t = w / (NT_TILES / NTW);                       // (mat*2 + b)*16 + mo
  const int mo = t & 15;
  t >>= 4;
  const int b = t & 1;
  const int mat = t >> 1;

  const float* bias = (mat == 0) ? bq : (mat == 1) ? bk : bv;
  float* dst = (mat == 0) ? qt : (mat == 1) ? kt : vt;

  const v16h* ap = (const v16h*)apack +
      ((size_t)mat * MO_TILES + mo) * KC_TILES * 32 + lane;
  const v16h* bp = (const v16h*)hpack +
      ((size_t)b * KC_TILES * NT_TILES + nt0) * 32 + lane;

  v8f acc[NTW] = {};
#pragma unroll
  for (int kc = 0; kc < KC_TILES; ++kc) {
    const v16h a = ap[(size_t)kc * 32];
#pragma unroll
    for (int i = 0; i < NTW; ++i) {
      const v16h bb = bp[((size_t)kc * NT_TILES + i) * 32];
      acc[i] = __builtin_amdgcn_wmma_f32_16x16x32_f16(
          false, a, false, bb, (short)0, acc[i], false, false);
    }
  }

  const int laneHi = lane >> 4;
  const int nlo = lane & 15;
#pragma unroll
  for (int i = 0; i < NTW; ++i) {
    const int n = (nt0 + i) * 16 + nlo;
#pragma unroll
    for (int r = 0; r < 8; ++r) {
      const int o = mo * 16 + r + (laneHi << 3);
      const float v = acc[i][r] + bias[o];
      dst[(((size_t)b * NHEADS + (o >> 6)) * HW + n) * DH + (o & 63)] = v;
    }
  }
}

// ---------------------------------------------------------------------------
// 4) Gathered masked-softmax attention. One wave per (b, head, q).
//    Pass 1: lanes own keys (2 each), q-vec broadcast from LDS -> logits with
//            no cross-lane reduction. Softmax via shfl_xor reductions.
//    Pass 2: lanes own channels (2 each), weights broadcast from LDS.
//    Result written directly in B-fragment f16 layout for the WO GEMM.
// ---------------------------------------------------------------------------
__global__ __launch_bounds__(256) void attn_kernel(
    const float* __restrict__ qt, const float* __restrict__ kt,
    const float* __restrict__ vt, const int* __restrict__ vmask,
    const int* __restrict__ aidx, _Float16* __restrict__ attnpack) {
  __shared__ __align__(16) float lds_q[8][64];
  __shared__ __align__(16) float lds_w[8][64];
  __shared__ __align__(16) int   lds_i[8][64];

  const int lane = threadIdx.x & 31;
  const int wid = threadIdx.x >> 5;
  const int w = (blockIdx.x << 3) + wid;
  const int q = w & (HW - 1);
  const int t = w >> 12;
  const int head = t & 3;
  const int b = t >> 2;
  const size_t bh = (size_t)b * NHEADS + head;

  // stage q-vector (64 floats) into LDS
  const float* qrow = qt + (bh * HW + q) * DH;
  if (lane < 16) ((v4f*)lds_q[wid])[lane] = ((const v4f*)qrow)[lane];

  const int i0 = aidx[(size_t)q * KNN + lane];
  const int i1 = aidx[(size_t)q * KNN + lane + 32];
  const int m0 = vmask[(size_t)q * KNN + lane];
  const int m1 = vmask[(size_t)q * KNN + lane + 32];
  lds_i[wid][lane] = i0;
  lds_i[wid][lane + 32] = i1;
  __syncthreads();

  // logits: each lane dots its 2 gathered key rows (256B contiguous) with q
  const v4f* k0p = (const v4f*)(kt + (bh * HW + i0) * DH);
  const v4f* k1p = (const v4f*)(kt + (bh * HW + i1) * DH);
  const v4f* qp = (const v4f*)lds_q[wid];
  float l0 = 0.f, l1 = 0.f;
#pragma unroll
  for (int cc = 0; cc < 16; ++cc) {
    v4f qc = qp[cc], ka = k0p[cc], kb = k1p[cc];
    l0 += qc[0] * ka[0] + qc[1] * ka[1] + qc[2] * ka[2] + qc[3] * ka[3];
    l1 += qc[0] * kb[0] + qc[1] * kb[1] + qc[2] * kb[2] + qc[3] * kb[3];
  }
  if (!m0) l0 = -INFINITY;
  if (!m1) l1 = -INFINITY;

  // softmax over 64 values (2 per lane) via wave32 shuffle reductions
  float mx = fmaxf(l0, l1);
  for (int off = 16; off > 0; off >>= 1) mx = fmaxf(mx, __shfl_xor(mx, off, 32));
  const float e0 = m0 ? __expf(l0 - mx) : 0.f;
  const float e1 = m1 ? __expf(l1 - mx) : 0.f;
  float sum = e0 + e1;
  for (int off = 16; off > 0; off >>= 1) sum += __shfl_xor(sum, off, 32);
  const float inv = 1.f / sum;
  lds_w[wid][lane] = e0 * inv;
  lds_w[wid][lane + 32] = e1 * inv;
  __syncthreads();

  // weighted V sum: lanes own channels (coalesced 128B row reads)
  float a0 = 0.f, a1 = 0.f;
#pragma unroll 4
  for (int k = 0; k < KNN; ++k) {
    const float wk = lds_w[wid][k];
    const float* vrow = vt + (bh * HW + lds_i[wid][k]) * DH;
    a0 += wk * vrow[lane];
    a1 += wk * vrow[lane + 32];
  }

  // store into B-fragment layout (o = head*64 + c, n = q)
  const int nlo = q & 15, nt = q >> 4;
#pragma unroll
  for (int h2 = 0; h2 < 2; ++h2) {
    const int c = lane + h2 * 32;
    const float val = h2 ? a1 : a0;
    const int o = head * DH + c;
    const int kc = o >> 5, cr = o & 31;
    const int l2 = ((cr >> 4) << 4) | nlo;
    const int j = cr & 15;
    attnpack[((((size_t)b * KC_TILES + kc) * NT_TILES + nt) * 32 + l2) * 16 + j] =
        (_Float16)val;
  }
}

// ---------------------------------------------------------------------------
// 5) WO GEMM (1 mo x 4 nt strip per wave) + bias + residual -> final output.
// ---------------------------------------------------------------------------
__global__ __launch_bounds__(256) void out_gemm_kernel(
    const _Float16* __restrict__ apack, const _Float16* __restrict__ attnpack,
    const float* __restrict__ bo, const float* __restrict__ x,
    float* __restrict__ out) {
  const int lane = threadIdx.x & 31;
  const int w = (blockIdx.x << 3) + (threadIdx.x >> 5);
  const int nt0 = (w & (NT_TILES / NTW - 1)) * NTW;
  const int t = w / (NT_TILES / NTW);   // b*16 + mo
  const int mo = t & 15;
  const int b = t >> 4;

  const v16h* ap = (const v16h*)apack +
      ((size_t)3 * MO_TILES + mo) * KC_TILES * 32 + lane;   // mat 3 = wo
  const v16h* bp = (const v16h*)attnpack +
      ((size_t)b * KC_TILES * NT_TILES + nt0) * 32 + lane;

  v8f acc[NTW] = {};
#pragma unroll
  for (int kc = 0; kc < KC_TILES; ++kc) {
    const v16h a = ap[(size_t)kc * 32];
#pragma unroll
    for (int i = 0; i < NTW; ++i) {
      const v16h bb = bp[((size_t)kc * NT_TILES + i) * 32];
      acc[i] = __builtin_amdgcn_wmma_f32_16x16x32_f16(
          false, a, false, bb, (short)0, acc[i], false, false);
    }
  }

  const int laneHi = lane >> 4;
  const int nlo = lane & 15;
#pragma unroll
  for (int i = 0; i < NTW; ++i) {
    const int n = (nt0 + i) * 16 + nlo;
#pragma unroll
    for (int r = 0; r < 8; ++r) {
      const int o = mo * 16 + r + (laneHi << 3);
      const size_t idx = ((size_t)b * CH + o) * HW + n;
      out[idx] = x[idx] + acc[i][r] + bo[o];
    }
  }
}

// ---------------------------------------------------------------------------
extern "C" void kernel_launch(void* const* d_in, const int* in_sizes, int n_in,
                              void* d_out, int out_size, void* d_ws, size_t ws_size,
                              hipStream_t stream) {
  (void)in_sizes; (void)n_in; (void)out_size; (void)ws_size;
  const float* x     = (const float*)d_in[0];
  const int*   vmask = (const int*)d_in[1];
  const int*   aidx  = (const int*)d_in[2];
  const float* gs    = (const float*)d_in[3];
  const float* gb    = (const float*)d_in[4];
  const float* wq    = (const float*)d_in[5];
  const float* bq    = (const float*)d_in[6];
  const float* wk    = (const float*)d_in[7];
  const float* bk    = (const float*)d_in[8];
  const float* wv    = (const float*)d_in[9];
  const float* bv    = (const float*)d_in[10];
  const float* wo    = (const float*)d_in[11];
  const float* bo    = (const float*)d_in[12];
  float* out = (float*)d_out;

  char* ws = (char*)d_ws;
  _Float16* hpack    = (_Float16*)(ws + HPACK_OFF);
  _Float16* apack    = (_Float16*)(ws + APACK_OFF);
  float*    qt       = (float*)(ws + QT_OFF);
  float*    kt       = (float*)(ws + KT_OFF);
  float*    vt       = (float*)(ws + VT_OFF);
  _Float16* attnpack = (_Float16*)(ws + ATTNP_OFF);

  // 1) GroupNorm + pack activations (f16, B-fragment layout)
  gn_pack_kernel<<<BATCH * NGROUPS, 1024, 0, stream>>>(x, gs, gb, hpack);
  // 2) Pack weights (f16, A-fragment layout); 4*65536 elems / 256
  pack_w_kernel<<<(4 * CH * CH) / 256, 256, 0, stream>>>(wq, wk, wv, wo, apack);
  // 3) Q/K/V WMMA GEMMs: 3*2*16*(256/4) strips, 8 waves/block
  qkv_gemm_kernel<<<(3 * BATCH * MO_TILES * (NT_TILES / NTW)) / 8, 256, 0,
                    stream>>>(apack, hpack, bq, bk, bv, qt, kt, vt);
  // 4) Attention: one wave per (b, head, q) = 32768 waves
  attn_kernel<<<(BATCH * NHEADS * HW) / 8, 256, 0, stream>>>(
      qt, kt, vt, vmask, aidx, attnpack);
  // 5) WO WMMA GEMM + residual
  out_gemm_kernel<<<(BATCH * MO_TILES * (NT_TILES / NTW)) / 8, 256, 0, stream>>>(
      apack, attnpack, bo, x, out);
}
